// MoELayer_83932250899013
// MI455X (gfx1250) — compile-verified, matrix operand order checked by reading, not run on hardware
//
#include <hip/hip_runtime.h>
#include <hip/hip_bf16.h>
#include <math.h>

#define HIDDEN 1024
#define FFN    4096
#define NEXP   8
#define NTOK   1024
#define TM     16
#define FSPLIT 2                 // blocks per (expert, token-tile): each does 16 of 32 F-chunks

typedef __attribute__((ext_vector_type(16))) __bf16 v16bf;
typedef __attribute__((ext_vector_type(8)))  float  v8f;

union AB { v16bf v; __bf16 e[16]; uint4 q[2]; };

#define XS_STRIDE (HIDDEN + 8)   // bf16 elems, padded to spread LDS banks
#define HS_STRIDE (128 + 8)

#if defined(__has_builtin)
#if __has_builtin(__builtin_amdgcn_tensor_load_to_lds) && __has_builtin(__builtin_amdgcn_s_wait_tensorcnt)
#define HAVE_TDM 1
#endif
#endif

#ifdef HAVE_TDM
typedef unsigned int u32x4 __attribute__((ext_vector_type(4)));
typedef int          i32x4 __attribute__((ext_vector_type(4)));
typedef int          i32x8 __attribute__((ext_vector_type(8)));
#endif

// ---------------- zero output + routing counters ----------------
__global__ void k_zero(float* __restrict__ out, int n, int* __restrict__ counts) {
  int i = blockIdx.x * blockDim.x + threadIdx.x;
  int stride = gridDim.x * blockDim.x;
  for (int j = i; j < n; j += stride) out[j] = 0.f;
  if (blockIdx.x == 0 && threadIdx.x < NEXP) counts[threadIdx.x] = 0;
}

// ---------------- router: 1 wave per token ----------------
__global__ __launch_bounds__(32) void k_router(
    const float* __restrict__ x, const float* __restrict__ gw,
    const float* __restrict__ gb, int* __restrict__ counts,
    int* __restrict__ tok_id, float* __restrict__ tok_w) {
  const int t = blockIdx.x;
  const int lane = threadIdx.x;
  const float* xr = x + (size_t)t * HIDDEN;
  float p[NEXP];
#pragma unroll
  for (int e = 0; e < NEXP; ++e) p[e] = 0.f;
  for (int i = lane; i < HIDDEN; i += 32) {
    float xv = xr[i];
#pragma unroll
    for (int e = 0; e < NEXP; ++e) p[e] += xv * gw[e * HIDDEN + i];
  }
#pragma unroll
  for (int e = 0; e < NEXP; ++e) {
#pragma unroll
    for (int off = 16; off > 0; off >>= 1) p[e] += __shfl_xor(p[e], off, 32);
  }
  if (lane == 0) {
#pragma unroll
    for (int e = 0; e < NEXP; ++e) p[e] += gb[e];
    int i1 = 0; float l1 = p[0];
#pragma unroll
    for (int e = 1; e < NEXP; ++e) if (p[e] > l1) { l1 = p[e]; i1 = e; }
    int i2 = (i1 == 0) ? 1 : 0; float l2 = p[i2];
#pragma unroll
    for (int e = 0; e < NEXP; ++e)
      if (e != i1 && e != i2 && p[e] > l2) { l2 = p[e]; i2 = e; }
    float e2 = __expf(l2 - l1);
    float w1 = 1.f / (1.f + e2);
    float w2 = e2 * w1;
    int s1 = atomicAdd(&counts[i1], 1);
    tok_id[i1 * NTOK + s1] = t; tok_w[i1 * NTOK + s1] = w1;
    int s2 = atomicAdd(&counts[i2], 1);
    tok_id[i2 * NTOK + s2] = t; tok_w[i2 * NTOK + s2] = w2;
  }
}

// ---------------- pack fp32 weights into WMMA-B fragment-ready bf16 ----------------
__global__ __launch_bounds__(256) void k_pack(const float* __restrict__ w1,
                                              const float* __restrict__ w2,
                                              __bf16* __restrict__ w1f,
                                              __bf16* __restrict__ w2f) {
  int bx = blockIdx.x;
  const float* src; __bf16* dst; int K, N;
  if (bx < 8192) { src = w1; dst = w1f; K = HIDDEN; N = FFN; }
  else { bx -= 8192; src = w2; dst = w2f; K = FFN; N = HIDDEN; }
  const int ntg_per = N / 128;
  const int per_e = (K / 32) * ntg_per;
  const int e = bx / per_e;
  const int r = bx % per_e;
  const int kk  = r / ntg_per;
  const int ntg = r % ntg_per;
  const int kr  = threadIdx.x & 31;
  const int ntl = threadIdx.x >> 5;
  const int nt  = ntg * 8 + ntl;
  const int k   = kk * 32 + kr;
  const float* s = src + (size_t)e * K * N + (size_t)k * N + nt * 16;
  const int l   = ((kr >> 3) & 1) << 4;
  const int pos = (kr & 7) + ((kr >> 4) << 3);
  __bf16* d = dst + (size_t)e * K * N + ((size_t)(nt * (K / 32) + kk)) * 512 + pos;
#pragma unroll
  for (int j = 0; j < 4; ++j) {
    float4 v = *(const float4*)(s + j * 4);
    d[(l + j * 4 + 0) * 16] = (__bf16)v.x;
    d[(l + j * 4 + 1) * 16] = (__bf16)v.y;
    d[(l + j * 4 + 2) * 16] = (__bf16)v.z;
    d[(l + j * 4 + 3) * 16] = (__bf16)v.w;
  }
}

// ---------------- fused expert FFN ----------------
template <bool FRAG>
__global__ __launch_bounds__(256) void k_ffn(
    const float* __restrict__ x,
    const float* __restrict__ w1, const float* __restrict__ b1,
    const float* __restrict__ w2, const float* __restrict__ b2,
    const __bf16* __restrict__ w1f, const __bf16* __restrict__ w2f,
    const int* __restrict__ counts, const int* __restrict__ tok_id,
    const float* __restrict__ tok_w, float* __restrict__ out) {
  __shared__ __align__(16) __bf16 xs[TM * XS_STRIDE];
  __shared__ __align__(16) __bf16 hs[2 * TM * HS_STRIDE];   // double-buffered
  __shared__ int   s_tok[TM];
  __shared__ float s_w[TM];

  const int e    = blockIdx.x >> 7;          // 64 tiles * FSPLIT per expert
  const int rest = blockIdx.x & 127;
  const int tile = rest >> 1;
  const int half = rest & 1;                 // which 16 of the 32 F-chunks
  const int cnt  = counts[e];
  if (tile * TM >= cnt) return;              // uniform: EXEC stays all-ones

  const int tid  = threadIdx.x;
  const int wave = tid >> 5;
  const int lane = tid & 31;
  const int l16  = lane & 15;
  const int hi8  = (lane >> 4) << 3;

  if (tid < TM) {
    int slot = tile * TM + tid;
    if (slot < cnt) { s_tok[tid] = tok_id[e * NTOK + slot]; s_w[tid] = tok_w[e * NTOK + slot]; }
    else            { s_tok[tid] = 0;                       s_w[tid] = 0.f; }
  }
  __syncthreads();

#ifdef HAVE_TDM
  { // TDM gather-mode: DMA 16 token rows (fp32) into LDS, then convert to bf16
    __shared__ __align__(16) float xf[TM * HIDDEN];
    if (tid == 0) {
      unsigned lds = (unsigned)(size_t)(void*)&xf[0];
      unsigned long long ga = (unsigned long long)(size_t)x;
      u32x4 g0 = { 0x80000001u,                      // count=1, gather_mode=1, 16-bit idx
                   lds,
                   (unsigned)(ga & 0xffffffffu),
                   (unsigned)((ga >> 32) & 0x01ffffffu) | (2u << 30) }; // type=2
      i32x8 g1 = { (int)(2u << 16),                  // data_size=4B
                   (int)(HIDDEN << 16),              // tensor_dim0 lo16
                   (int)(NTOK << 16),                // dim0 hi=0 | tensor_dim1 lo16
                   (int)(HIDDEN << 16),              // dim1 hi=0 | tile_dim0
                   (int)TM,                          // tile_dim1 = #indices
                   (int)HIDDEN,                      // tensor_dim0_stride lo32
                   0, 0 };
      i32x4 g2 = { s_tok[0] | (s_tok[1] << 16),  s_tok[2] | (s_tok[3] << 16),
                   s_tok[4] | (s_tok[5] << 16),  s_tok[6] | (s_tok[7] << 16) };
      i32x4 g3 = { s_tok[8] | (s_tok[9] << 16),  s_tok[10] | (s_tok[11] << 16),
                   s_tok[12] | (s_tok[13] << 16), s_tok[14] | (s_tok[15] << 16) };
#if __clang_major__ >= 23
      __builtin_amdgcn_tensor_load_to_lds(g0, g1, g2, g3, (i32x8){0,0,0,0,0,0,0,0}, 0);
#else
      __builtin_amdgcn_tensor_load_to_lds(g0, g1, g2, g3, 0);
#endif
    }
    __builtin_amdgcn_s_wait_tensorcnt(0);
    __syncthreads();
#pragma unroll
    for (int i = 0; i < (TM * HIDDEN) / (256 * 4); ++i) {  // 16 float4 per thread
      int idx = (tid + i * 256) * 4;
      float4 v = *(const float4*)&xf[idx];
      int row = idx >> 10, col = idx & (HIDDEN - 1);
      __bf16* d = &xs[row * XS_STRIDE + col];
      d[0] = (__bf16)v.x; d[1] = (__bf16)v.y; d[2] = (__bf16)v.z; d[3] = (__bf16)v.w;
    }
  }
#else
  { // fallback: per-thread gather of token rows
    int row = tid >> 4, sub = tid & 15;
    const float* xr = x + (size_t)s_tok[row] * HIDDEN;
#pragma unroll
    for (int i = 0; i < HIDDEN / (16 * 4); ++i) {
      int c = (sub + i * 16) * 4;
      float4 v = *(const float4*)(xr + c);
      __bf16* d = &xs[row * XS_STRIDE + c];
      d[0] = (__bf16)v.x; d[1] = (__bf16)v.y; d[2] = (__bf16)v.z; d[3] = (__bf16)v.w;
    }
  }
#endif
  __syncthreads();

  const float* w1e = w1 + (size_t)e * HIDDEN * FFN;
  const float* w2e = w2 + (size_t)e * FFN * HIDDEN;
  const float* b1e = b1 + (size_t)e * FFN;
  const float* b2e = b2 + (size_t)e * HIDDEN;
  const __bf16* w1fe = w1f + (size_t)e * HIDDEN * FFN;
  const __bf16* w2fe = w2f + (size_t)e * FFN * HIDDEN;

  v8f yacc[8];
#pragma unroll
  for (int i = 0; i < 8; ++i)
    yacc[i] = (v8f){0.f, 0.f, 0.f, 0.f, 0.f, 0.f, 0.f, 0.f};

  const int ncol2   = wave * 128;
  const int fc_lo   = half * (FFN / 128 / FSPLIT);
  const int fc_hi   = fc_lo + (FFN / 128 / FSPLIT);

  for (int fc = fc_lo; fc < fc_hi; ++fc) {
    __bf16* hbuf = &hs[(fc & 1) * TM * HS_STRIDE];
    // ---- GEMM1: this wave's 16x16 h-tile at columns fc*128 + wave*16 ----
    const int n1 = fc * 128 + wave * 16 + l16;
    v8f acc1 = (v8f){0.f, 0.f, 0.f, 0.f, 0.f, 0.f, 0.f, 0.f};
    if (FRAG) {
      const __bf16* bp = w1fe + ((size_t)((fc * 8 + wave) * 32)) * 512 + lane * 16;
      __builtin_prefetch(w2fe + ((size_t)(wave * 8 * 128 + fc * 4)) * 512 + lane * 16, 0, 1);
      for (int kk = 0; kk < HIDDEN / 32; ++kk) {
        AB a, b;
        a.q[0] = *(const uint4*)&xs[l16 * XS_STRIDE + kk * 32 + hi8];
        a.q[1] = *(const uint4*)&xs[l16 * XS_STRIDE + kk * 32 + 16 + hi8];
        b.q[0] = *(const uint4*)(bp);
        b.q[1] = *(const uint4*)(bp + 8);
        bp += 512;
        acc1 = __builtin_amdgcn_wmma_f32_16x16x32_bf16(false, a.v, false, b.v,
                                                       (short)0, acc1, false, false);
      }
    } else {
      for (int kk = 0; kk < HIDDEN / 32; ++kk) {
        const int kb = kk * 32;
        AB a, b;
        a.q[0] = *(const uint4*)&xs[l16 * XS_STRIDE + kb + hi8];
        a.q[1] = *(const uint4*)&xs[l16 * XS_STRIDE + kb + 16 + hi8];
#pragma unroll
        for (int i = 0; i < 8; ++i) {
          b.e[i]     = (__bf16)w1e[(size_t)(kb + hi8 + i) * FFN + n1];
          b.e[i + 8] = (__bf16)w1e[(size_t)(kb + 16 + hi8 + i) * FFN + n1];
        }
        acc1 = __builtin_amdgcn_wmma_f32_16x16x32_bf16(false, a.v, false, b.v,
                                                       (short)0, acc1, false, false);
      }
    }
    { // bias + exact GeLU -> hbuf (row-major 16 x 128 bf16)
      float bias = b1e[n1];
#pragma unroll
      for (int r = 0; r < 8; ++r) {
        int m = r + hi8;
        float v = acc1[r] + bias;
        float g = 0.5f * v * (1.f + erff(v * 0.70710678118654752440f));
        hbuf[m * HS_STRIDE + wave * 16 + l16] = (__bf16)g;
      }
    }
    __syncthreads();   // single barrier per chunk: double-buffered hs

    // ---- GEMM2: yacc += h_chunk(16x128) @ W2[fc*128.., ncol2..+128] ----
    AB a4[4];
#pragma unroll
    for (int ks = 0; ks < 4; ++ks) {
      a4[ks].q[0] = *(const uint4*)&hbuf[l16 * HS_STRIDE + ks * 32 + hi8];
      a4[ks].q[1] = *(const uint4*)&hbuf[l16 * HS_STRIDE + ks * 32 + 16 + hi8];
    }
#pragma unroll
    for (int nt = 0; nt < 8; ++nt) {
      const int n2 = ncol2 + nt * 16 + l16;
      if (FRAG) {
        const __bf16* bp = w2fe + ((size_t)((wave * 8 + nt) * 128 + fc * 4)) * 512 + lane * 16;
#pragma unroll
        for (int ks = 0; ks < 4; ++ks) {
          AB b;
          b.q[0] = *(const uint4*)(bp);
          b.q[1] = *(const uint4*)(bp + 8);
          bp += 512;
          yacc[nt] = __builtin_amdgcn_wmma_f32_16x16x32_bf16(false, a4[ks].v, false, b.v,
                                                             (short)0, yacc[nt], false, false);
        }
      } else {
#pragma unroll
        for (int ks = 0; ks < 4; ++ks) {
          const int kb = fc * 128 + ks * 32;
          AB b;
#pragma unroll
          for (int i = 0; i < 8; ++i) {
            b.e[i]     = (__bf16)w2e[(size_t)(kb + hi8 + i) * HIDDEN + n2];
            b.e[i + 8] = (__bf16)w2e[(size_t)(kb + 16 + hi8 + i) * HIDDEN + n2];
          }
          yacc[nt] = __builtin_amdgcn_wmma_f32_16x16x32_bf16(false, a4[ks].v, false, b.v,
                                                             (short)0, yacc[nt], false, false);
        }
      }
    }
  }

  // ---- epilogue: +b2 (half 0 only), scale by router weight, scatter-add ----
#pragma unroll
  for (int nt = 0; nt < 8; ++nt) {
    const int n2 = ncol2 + nt * 16 + l16;
    const float bias = (half == 0) ? b2e[n2] : 0.f;
#pragma unroll
    for (int r = 0; r < 8; ++r) {
      int m = r + hi8;
      float wgt = s_w[m];
      if (wgt != 0.f) {
        float v = yacc[nt][r] + bias;
        unsafeAtomicAdd(&out[(size_t)s_tok[m] * HIDDEN + n2], wgt * v);
      }
    }
  }
}

extern "C" void kernel_launch(void* const* d_in, const int* in_sizes, int n_in,
                              void* d_out, int out_size, void* d_ws, size_t ws_size,
                              hipStream_t stream) {
  const float* x  = (const float*)d_in[0];
  const float* gw = (const float*)d_in[1];
  const float* gb = (const float*)d_in[2];
  const float* w1 = (const float*)d_in[3];
  const float* b1 = (const float*)d_in[4];
  const float* w2 = (const float*)d_in[5];
  const float* b2 = (const float*)d_in[6];
  float* out = (float*)d_out;

  // ws layout: counters | token ids | token weights | (optional) packed bf16 weights
  int*   counts = (int*)d_ws;
  int*   tok_id = (int*)((char*)d_ws + 256);
  float* tok_w  = (float*)((char*)d_ws + 256 + NEXP * NTOK * sizeof(int));
  const size_t frag_off   = 256 + 2 * (size_t)NEXP * NTOK * 4 + 256;      // ~66 KB
  const size_t frag_bytes = 2ull * NEXP * (size_t)HIDDEN * FFN * 2;       // 128 MB bf16
  __bf16* w1f = (__bf16*)((char*)d_ws + frag_off);
  __bf16* w2f = w1f + (size_t)NEXP * HIDDEN * FFN;
  const bool packed = ws_size >= frag_off + frag_bytes;

  const int ffn_blocks = NEXP * (NTOK / TM) * FSPLIT;   // 1024

  k_zero<<<256, 256, 0, stream>>>(out, out_size, counts);
  k_router<<<NTOK, 32, 0, stream>>>(x, gw, gb, counts, tok_id, tok_w);
  if (packed) {
    k_pack<<<16384, 256, 0, stream>>>(w1, w2, w1f, w2f);
    k_ffn<true><<<ffn_blocks, 256, 0, stream>>>(x, w1, b1, w2, b2, w1f, w2f,
                                                counts, tok_id, tok_w, out);
  } else {
    k_ffn<false><<<ffn_blocks, 256, 0, stream>>>(x, w1, b1, w2, b2, w1f, w2f,
                                                 counts, tok_id, tok_w, out);
  }
}
